// QuantizedLinear_5446018531482
// MI455X (gfx1250) — 8-bit matrix operand order repaired and checked by reading, compiled, not run on hardware
//
#include <hip/hip_runtime.h>
#include <stdint.h>

typedef __attribute__((ext_vector_type(8))) int v8i;
typedef __attribute__((ext_vector_type(4))) int v4i;
typedef __attribute__((ext_vector_type(2))) int v2i;

#define QMAXF 127.0f
#define EPSF  1e-8f

#define B_DIM 4
#define S_DIM 2048
#define K_DIM 4096
#define N_DIM 4096
#define M_DIM (B_DIM * S_DIM)


__global__ __launch_bounds__(32) void ql_init_amax(unsigned int* amax) {
    if (threadIdx.x < 8) amax[threadIdx.x] = 0u;
}

__global__ __launch_bounds__(256) void ql_reduce_amax(
    const float4* __restrict__ src, size_t count4_per_slice,
    unsigned int* __restrict__ amax, int dst_base) {
    __shared__ float smax[256];
    const float4* p = src + (size_t)blockIdx.y * count4_per_slice;
    float m = 0.0f;
    for (size_t i = (size_t)blockIdx.x * blockDim.x + threadIdx.x;
         i < count4_per_slice;
         i += (size_t)gridDim.x * blockDim.x) {
        float4 v = p[i];
        m = fmaxf(m, fmaxf(fmaxf(fabsf(v.x), fabsf(v.y)),
                           fmaxf(fabsf(v.z), fabsf(v.w))));
    }
    smax[threadIdx.x] = m;
    __syncthreads();
    for (int s = 128; s > 0; s >>= 1) {
        if ((int)threadIdx.x < s)
            smax[threadIdx.x] = fmaxf(smax[threadIdx.x], smax[threadIdx.x + s]);
        __syncthreads();
    }
    if (threadIdx.x == 0)
        atomicMax(&amax[dst_base + blockIdx.y], __float_as_uint(smax[0]));
}

__device__ __forceinline__ int ql_q8(float x, float inv) {
    float r = rintf(x * inv);
    r = fminf(fmaxf(r, -QMAXF), QMAXF);
    return (int)r;
}

__global__ __launch_bounds__(256) void ql_quant_x(
    const float4* __restrict__ x, unsigned int* __restrict__ q,
    const unsigned int* __restrict__ amax, size_t count4) {
    for (size_t i = (size_t)blockIdx.x * blockDim.x + threadIdx.x;
         i < count4;
         i += (size_t)gridDim.x * blockDim.x) {
        int b = (int)(i >> 21);
        float a = fmaxf(__uint_as_float(amax[b]), EPSF);
        float inv = QMAXF / a;
        float4 v = x[i];
        unsigned int p = ((unsigned)(ql_q8(v.x, inv) & 0xff))
                       | ((unsigned)(ql_q8(v.y, inv) & 0xff) << 8)
                       | ((unsigned)(ql_q8(v.z, inv) & 0xff) << 16)
                       | ((unsigned)(ql_q8(v.w, inv) & 0xff) << 24);
        q[i] = p;
    }
}

__global__ __launch_bounds__(256) void ql_quant_w(
    const float4* __restrict__ w, unsigned int* __restrict__ q,
    const unsigned int* __restrict__ amax, size_t count4) {
    float a = fmaxf(__uint_as_float(amax[4]), EPSF);
    float inv = QMAXF / a;
    for (size_t i = (size_t)blockIdx.x * blockDim.x + threadIdx.x;
         i < count4;
         i += (size_t)gridDim.x * blockDim.x) {
        float4 v = w[i];
        unsigned int p = ((unsigned)(ql_q8(v.x, inv) & 0xff))
                       | ((unsigned)(ql_q8(v.y, inv) & 0xff) << 8)
                       | ((unsigned)(ql_q8(v.z, inv) & 0xff) << 16)
                       | ((unsigned)(ql_q8(v.w, inv) & 0xff) << 24);
        q[i] = p;
    }
}

__global__ __launch_bounds__(256) void ql_gemm_i8_wmma(
    const int8_t* __restrict__ xq, const int8_t* __restrict__ wq,
    const float* __restrict__ bias, const unsigned int* __restrict__ amax,
    float* __restrict__ out) {
    const int lane   = threadIdx.x & 31;
    const int wave   = threadIdx.x >> 5;
    const int wy     = wave >> 1;
    const int wx     = wave & 1;
    const int lane16 = lane & 15;
    const int hi     = lane >> 4;

    const int blockM = blockIdx.y * 128;
    const int blockN = blockIdx.x * 64;
    const int waveM  = blockM + wy * 32;
    const int waveN  = blockN + wx * 32;
    const int b      = blockM >> 11;

    v8i acc[2][2] = {};

    const int8_t* aptr0 = xq + (size_t)(waveM + lane16) * K_DIM + hi * 16;
    const int8_t* aptr1 = aptr0 + (size_t)16 * K_DIM;
    const int8_t* bptr0 = wq + (size_t)(waveN + lane16) * K_DIM + hi * 16;
    const int8_t* bptr1 = bptr0 + (size_t)16 * K_DIM;

#pragma unroll 2
    for (int k0 = 0; k0 < K_DIM; k0 += 64) {
        v2i t0, t1, t2, t3;
        v8i a0, a1, b0, b1;

        t0 = *(const v2i*)(aptr0 + k0);
        t1 = *(const v2i*)(aptr0 + k0 + 8);
        t2 = *(const v2i*)(aptr0 + k0 + 32);
        t3 = *(const v2i*)(aptr0 + k0 + 40);
        a0 = (v8i){t0.x, t0.y, t1.x, t1.y, t2.x, t2.y, t3.x, t3.y};

        t0 = *(const v2i*)(aptr1 + k0);
        t1 = *(const v2i*)(aptr1 + k0 + 8);
        t2 = *(const v2i*)(aptr1 + k0 + 32);
        t3 = *(const v2i*)(aptr1 + k0 + 40);
        a1 = (v8i){t0.x, t0.y, t1.x, t1.y, t2.x, t2.y, t3.x, t3.y};

        v4i u0 = *(const v4i*)(bptr0 + k0);
        v4i u1 = *(const v4i*)(bptr0 + k0 + 32);
        b0 = (v8i){u0.x, u0.y, u0.z, u0.w, u1.x, u1.y, u1.z, u1.w};

        u0 = *(const v4i*)(bptr1 + k0);
        u1 = *(const v4i*)(bptr1 + k0 + 32);
        b1 = (v8i){u0.x, u0.y, u0.z, u0.w, u1.x, u1.y, u1.z, u1.w};

        acc[0][0] = __builtin_amdgcn_wmma_i32_16x16x64_iu8(true, a0, true, b0, acc[0][0], false, false);
        acc[0][1] = __builtin_amdgcn_wmma_i32_16x16x64_iu8(true, a0, true, b1, acc[0][1], false, false);
        acc[1][0] = __builtin_amdgcn_wmma_i32_16x16x64_iu8(true, a1, true, b0, acc[1][0], false, false);
        acc[1][1] = __builtin_amdgcn_wmma_i32_16x16x64_iu8(true, a1, true, b1, acc[1][1], false, false);
    }

    const float ax = fmaxf(__uint_as_float(amax[b]), EPSF);
    const float aw = fmaxf(__uint_as_float(amax[4]), EPSF);
    const float oscale = (ax / QMAXF) * (aw / QMAXF);

#pragma unroll
    for (int i = 0; i < 2; ++i) {
#pragma unroll
        for (int j = 0; j < 2; ++j) {
            const int n = waveN + j * 16 + lane16;
            const float bn = bias[n];
#pragma unroll
            for (int r = 0; r < 8; ++r) {
                const int m = waveM + i * 16 + r + hi * 8;
                out[(size_t)m * N_DIM + n] = (float)acc[i][j][r] * oscale + bn;
            }
        }
    }
}

extern "C" void kernel_launch(void* const* d_in, const int* in_sizes, int n_in,
                              void* d_out, int out_size, void* d_ws, size_t ws_size,
                              hipStream_t stream) {
    const float* x    = (const float*)d_in[0];
    const float* w    = (const float*)d_in[1];
    const float* bias = (const float*)d_in[2];
    float* out        = (float*)d_out;

    unsigned int* amax = (unsigned int*)d_ws;
    int8_t* xq = (int8_t*)d_ws + 64;
    int8_t* wq = xq + (size_t)M_DIM * K_DIM;

    const size_t x_count4       = (size_t)M_DIM * K_DIM / 4;
    const size_t x_count4_per_b = (size_t)S_DIM * K_DIM / 4;
    const size_t w_count4       = (size_t)N_DIM * K_DIM / 4;

    ql_init_amax<<<1, 32, 0, stream>>>(amax);
    ql_reduce_amax<<<dim3(512, B_DIM), 256, 0, stream>>>(
        (const float4*)x, x_count4_per_b, amax, 0);
    ql_reduce_amax<<<dim3(1024, 1), 256, 0, stream>>>(
        (const float4*)w, w_count4, amax, 4);
    ql_quant_x<<<2048, 256, 0, stream>>>(
        (const float4*)x, (unsigned int*)xq, amax, x_count4);
    ql_quant_w<<<1024, 256, 0, stream>>>(
        (const float4*)w, (unsigned int*)wq, amax, w_count4);
    ql_gemm_i8_wmma<<<dim3(N_DIM / 64, M_DIM / 128), 256, 0, stream>>>(
        xq, wq, bias, amax, out);
}
